// QRNNLayer_67877663146083
// MI455X (gfx1250) — matrix operand order repaired and checked by reading, not compiled, not run on hardware
//
#include <hip/hip_runtime.h>

// ---------------------------------------------------------------------------
// QRNN layer for MI455X (gfx1250, wave32):
//   bf16 pre-pack of A (causal shift baked in) and W, then
//   gates GEMM (M=16384,N=3072,K=2048) via v_wmma_f32_16x16x32_bf16 with
//   double-buffered LDS stages (KC=64) and async global->LDS copies,
//   fused activations, segmented fo-pooling scan, fused h = O*c.
// ---------------------------------------------------------------------------

typedef __attribute__((ext_vector_type(16))) __bf16 v16bf;
typedef __attribute__((ext_vector_type(8)))  float  v8f;
typedef __attribute__((ext_vector_type(4)))  int    v4i;

#define B_   16
#define T_   1024
#define D_   1024
#define H_   1024
#define NT_  3072            // 3*H
#define KK_  2048            // D*K (k<1024 -> x[t-1], k>=1024 -> x[t])
#define M_   (B_*T_)         // 16384
#define BH_  (B_*H_)         // 16384 scan channels
#define SEG_ 8
#define TSEG_ (T_/SEG_)      // 128
#define KC_  64              // k-elements per LDS stage (2 wmma k-steps)
#define NSTAGE_ (KK_/KC_)    // 32

// ---- workspace layout (bytes) ----
#define WB_OFF   0ull
#define WB_BYTES (2ull*NT_*KK_)            // 12.58 MB bf16 packed weights [n][k]
#define AP_OFF   (WB_OFF + WB_BYTES)
#define AP_BYTES (2ull*M_*KK_)             // 64 MB bf16 packed activations [m][k]
#define Z_OFF    (AP_OFF + AP_BYTES)
#define ZF_BYTES (2ull*(size_t)T_*BH_)     // 32 MB each, bf16 [T][B][H]
#define F_OFF    (Z_OFF + ZF_BYTES)
#define P_OFF    (F_OFF + ZF_BYTES)
#define PC_BYTES (4ull*SEG_*BH_)           // 512 KB each, f32
#define C_OFF    (P_OFF + PC_BYTES)

// ---- CDNA5 async global->LDS path (guarded; falls back to VGPR staging) ----
#if defined(__has_builtin)
#if __has_builtin(__builtin_amdgcn_global_load_async_to_lds_b128)
#define USE_ASYNC_LDS 1
#endif
#if __has_builtin(__builtin_amdgcn_s_wait_asynccnt)
#define HAVE_WAIT_ASYNC_BUILTIN 1
#endif
#endif
#ifndef USE_ASYNC_LDS
#define USE_ASYNC_LDS 0
#endif

#if USE_ASYNC_LDS
// builtin signature: (v4i AS1*, v4i AS3*, imm offset, imm cpol)
#define GAS1(p) ((__attribute__((address_space(1))) v4i*)(unsigned long long)(p))
#define LAS3(p) ((__attribute__((address_space(3))) v4i*)(unsigned int)(unsigned long long)(p))
#if defined(HAVE_WAIT_ASYNC_BUILTIN)
#define WAIT_ASYNC() __builtin_amdgcn_s_wait_asynccnt(0)
#else
#define WAIT_ASYNC() asm volatile("s_wait_asynccnt 0x0" ::: "memory")
#endif
#endif

// ---------------------------------------------------------------------------
// Kernel 0: repack conv_w[n, d, kk] (fp32, kk innermost) -> Wb[n][d + 1024*kk]
// ---------------------------------------------------------------------------
__global__ __launch_bounds__(256) void pack_w_kernel(const float* __restrict__ w,
                                                     __bf16* __restrict__ wb) {
  const int idx = blockIdx.x * 256 + threadIdx.x;   // 0 .. NT_*D_-1
  const int n = idx >> 10;
  const int d = idx & 1023;
  const float2 v = *(const float2*)(w + (size_t)n * 2048 + 2 * d); // {w[n,d,0], w[n,d,1]}
  wb[(size_t)n * KK_ + d]        = (__bf16)v.x;
  wb[(size_t)n * KK_ + 1024 + d] = (__bf16)v.y;
}

// ---------------------------------------------------------------------------
// Kernel 1: pack A: Apack[b*T+t][0:1024]  = x[b,t-1,:] (0 for t==0)
//                   Apack[b*T+t][1024:]   = x[b,t,:]
// ---------------------------------------------------------------------------
__global__ __launch_bounds__(256) void pack_a_kernel(const float* __restrict__ x,
                                                     __bf16* __restrict__ ap) {
  const int idx = blockIdx.x * 256 + threadIdx.x;   // 0 .. M_*D_/8 - 1
  const int row = idx >> 7;            // b*T + t
  const int d0  = (idx & 127) << 3;    // 0..1016 step 8
  const float4* src = (const float4*)(x + (size_t)row * D_ + d0);
  const float4 v0 = src[0], v1 = src[1];
  union { __bf16 h[8]; uint4 q; } pk;
  pk.h[0] = (__bf16)v0.x; pk.h[1] = (__bf16)v0.y;
  pk.h[2] = (__bf16)v0.z; pk.h[3] = (__bf16)v0.w;
  pk.h[4] = (__bf16)v1.x; pk.h[5] = (__bf16)v1.y;
  pk.h[6] = (__bf16)v1.z; pk.h[7] = (__bf16)v1.w;
  *(uint4*)(ap + (size_t)row * KK_ + 1024 + d0) = pk.q;     // current slab
  const int t = row & 1023;
  if (t < T_ - 1) *(uint4*)(ap + (size_t)(row + 1) * KK_ + d0) = pk.q; // prev slab of row+1
  if (t == 0) {
    const uint4 z = {0u, 0u, 0u, 0u};
    *(uint4*)(ap + (size_t)row * KK_ + d0) = z;             // zero-pad t=0
  }
}

// ---------------------------------------------------------------------------
// Kernel 2: gates GEMM + bias + activations.
//   grid = (NT_/128, M_/128), block = 256 (8 waves).
//   Wave computes 64x32 = 4x2 tiles of 16x16 f32 accumulators.
//   Double-buffered LDS, KC=64 per stage, async global->LDS when available.
// ---------------------------------------------------------------------------
__global__ __launch_bounds__(256) void gemm_gates_kernel(
    const __bf16* __restrict__ ap, const __bf16* __restrict__ wb,
    const float* __restrict__ bias,
    __bf16* __restrict__ zws, __bf16* __restrict__ fws,
    float* __restrict__ outh)
{
  __shared__ __align__(16) __bf16 As[2][128 * KC_];   // [m][k], 16KB per buffer
  __shared__ __align__(16) __bf16 Bs[2][128 * KC_];   // [n][k], 16KB per buffer

  const int tid   = threadIdx.x;
  const int lane  = tid & 31;
  const int wave  = tid >> 5;
  const int waveM = wave & 1;     // 2 waves over M (64 rows each)
  const int waveN = wave >> 1;    // 4 waves over N (32 cols each)
  const int m0 = blockIdx.y * 128;
  const int n0 = blockIdx.x * 128;

  // cooperative-load coords: 2 threads per row, 32 bf16 (64B) each
  const int ldRow = tid >> 1;           // 0..127
  const int ldCol = (tid & 1) * 32;     // 0 or 32
  const __bf16* aSrc = ap + (size_t)(m0 + ldRow) * KK_ + ldCol;
  const __bf16* bSrc = wb + (size_t)(n0 + ldRow) * KK_ + ldCol;
  const int ldOff = ldRow * KC_ + ldCol;

  const v8f vzero = {};
  v8f acc[4][2];
  #pragma unroll
  for (int i = 0; i < 4; ++i)
    #pragma unroll
    for (int j = 0; j < 2; ++j)
      acc[i][j] = vzero;

  const int lhalf = lane >> 4;   // 0: lanes 0-15, 1: lanes 16-31
  const int lrow  = lane & 15;

  // ---- stage loader ----
#if USE_ASYNC_LDS
  #define LOAD_STAGE(buf, kbase)                                               \
    do {                                                                       \
      _Pragma("unroll")                                                        \
      for (int q = 0; q < 4; ++q) {                                            \
        __builtin_amdgcn_global_load_async_to_lds_b128(                        \
            GAS1(aSrc + (kbase) + q * 8), LAS3(&As[buf][ldOff + q * 8]), 0, 0);\
        __builtin_amdgcn_global_load_async_to_lds_b128(                        \
            GAS1(bSrc + (kbase) + q * 8), LAS3(&Bs[buf][ldOff + q * 8]), 0, 0);\
      }                                                                        \
    } while (0)
  #define STAGE_FENCE() WAIT_ASYNC()
#else
  uint4 ra[4], rb[4];
  #define LOAD_STAGE_FETCH(kbase)                                              \
    do {                                                                       \
      _Pragma("unroll")                                                        \
      for (int q = 0; q < 4; ++q) {                                            \
        ra[q] = *(const uint4*)(aSrc + (kbase) + q * 8);                       \
        rb[q] = *(const uint4*)(bSrc + (kbase) + q * 8);                       \
      }                                                                        \
    } while (0)
  #define LOAD_STAGE_COMMIT(buf)                                               \
    do {                                                                       \
      _Pragma("unroll")                                                        \
      for (int q = 0; q < 4; ++q) {                                            \
        *(uint4*)&As[buf][ldOff + q * 8] = ra[q];                              \
        *(uint4*)&Bs[buf][ldOff + q * 8] = rb[q];                              \
      }                                                                        \
    } while (0)
#endif

  // ---- prologue: stage 0 ----
#if USE_ASYNC_LDS
  LOAD_STAGE(0, 0);
  STAGE_FENCE();
#else
  LOAD_STAGE_FETCH(0);
  LOAD_STAGE_COMMIT(0);
#endif
  __syncthreads();

  for (int kc = 0; kc < NSTAGE_; ++kc) {
    const int cur = kc & 1;
    const int nxt = cur ^ 1;
    // issue next-stage loads (overlap with WMMA below)
#if USE_ASYNC_LDS
    if (kc + 1 < NSTAGE_) LOAD_STAGE(nxt, (kc + 1) * KC_);
#else
    if (kc + 1 < NSTAGE_) LOAD_STAGE_FETCH((kc + 1) * KC_);
#endif

    // ---- compute: 2 k-steps of 32, 8 wmma each ----
    #pragma unroll
    for (int ks = 0; ks < 2; ++ks) {
      const int kb = ks * 32;
      // B fragments: lane holds column n=lrow, K = lhalf*16 .. +15 (contig)
      v16bf bfrag[2];
      #pragma unroll
      for (int j = 0; j < 2; ++j) {
        const int nr = waveN * 32 + j * 16 + lrow;
        union { uint4 q[2]; v16bf v; } u;
        const uint4* p = (const uint4*)&Bs[cur][nr * KC_ + kb + lhalf * 16];
        u.q[0] = p[0]; u.q[1] = p[1];
        bfrag[j] = u.v;
      }
      // A fragments: lane holds row m=lrow, K = lhalf*16 .. +15 (contig)
      #pragma unroll
      for (int i = 0; i < 4; ++i) {
        const int mr = waveM * 64 + i * 16 + lrow;
        union { uint4 q[2]; v16bf v; } u;
        u.q[0] = *(const uint4*)&As[cur][mr * KC_ + kb + lhalf * 16];
        u.q[1] = *(const uint4*)&As[cur][mr * KC_ + kb + lhalf * 16 + 8];
        const v16bf afrag = u.v;
        #pragma unroll
        for (int j = 0; j < 2; ++j) {
          acc[i][j] = __builtin_amdgcn_wmma_f32_16x16x32_bf16(
              false, afrag, false, bfrag[j], (short)0, acc[i][j], false, false);
        }
      }
    }

#if USE_ASYNC_LDS
    if (kc + 1 < NSTAGE_) STAGE_FENCE();   // own async writes to nxt done
#else
    if (kc + 1 < NSTAGE_) LOAD_STAGE_COMMIT(nxt);
#endif
    __syncthreads();
  }

  // ---- epilogue: bias + activation + scatter ----
  // C/D layout: lane l, vgpr v -> M = v + 8*(l>=16), N = l%16
  const int mbase = m0 + waveM * 64 + (lhalf << 3);
  #pragma unroll
  for (int j = 0; j < 2; ++j) {
    const int ncol = n0 + waveN * 32 + j * 16 + lrow;
    const float bv = bias[ncol];
    const int gate = ncol >> 10;      // 0=Z(tanh) 1=F(sig) 2=O(sig) — wave-uniform
    const int hidx = ncol & 1023;
    #pragma unroll
    for (int i = 0; i < 4; ++i) {
      #pragma unroll
      for (int v = 0; v < 8; ++v) {
        const int mrow = mbase + i * 16 + v;
        const int bb = mrow >> 10;
        const int tt = mrow & 1023;
        const float g = acc[i][j][v] + bv;
        if (gate == 0) {
          zws[(size_t)tt * BH_ + bb * H_ + hidx] = (__bf16)tanhf(g);
        } else if (gate == 1) {
          fws[(size_t)tt * BH_ + bb * H_ + hidx] = (__bf16)(1.0f / (1.0f + __expf(-g)));
        } else {
          outh[((size_t)bb * T_ + tt) * H_ + hidx] = 1.0f / (1.0f + __expf(-g));
        }
      }
    }
  }
}

// ---------------------------------------------------------------------------
// Kernel 3: per-segment reduce:  P = prod(f), Cseg = scan end with c_in = 0.
// ---------------------------------------------------------------------------
__global__ __launch_bounds__(256) void scan_seg_reduce(
    const __bf16* __restrict__ zws, const __bf16* __restrict__ fws,
    float* __restrict__ P, float* __restrict__ C)
{
  const int s = blockIdx.x >> 6;
  const int g = blockIdx.x & 63;
  const int b = g >> 2;
  const int h = ((g & 3) << 8) + threadIdx.x;
  const int chan = b * H_ + h;
  float c = 0.f, p = 1.f;
  const int t0 = s * TSEG_;
  for (int t = 0; t < TSEG_; ++t) {
    const size_t idx = (size_t)(t0 + t) * BH_ + chan;
    const float z = (float)zws[idx];
    const float f = (float)fws[idx];
    c = f * c + (1.f - f) * z;
    p *= f;
  }
  P[(size_t)s * BH_ + chan] = p;
  C[(size_t)s * BH_ + chan] = c;
}

// ---------------------------------------------------------------------------
// Kernel 4: carry propagation across segments; rewrites C[s] := carry-in of s.
// ---------------------------------------------------------------------------
__global__ __launch_bounds__(256) void scan_carry(
    const float* __restrict__ P, float* __restrict__ C)
{
  const int chan = blockIdx.x * 256 + threadIdx.x;
  float cin = 0.f;
  #pragma unroll
  for (int s = 0; s < SEG_; ++s) {
    const size_t idx = (size_t)s * BH_ + chan;
    const float p  = P[idx];
    const float ce = C[idx];
    C[idx] = cin;                  // carry-in for segment s
    cin = p * cin + ce;            // affine composition
  }
}

// ---------------------------------------------------------------------------
// Kernel 5: final pass — rerun segment scan from carry-in, write c and h=O*c.
// ---------------------------------------------------------------------------
__global__ __launch_bounds__(256) void scan_final(
    const __bf16* __restrict__ zws, const __bf16* __restrict__ fws,
    const float* __restrict__ Cin,
    float* __restrict__ outc, float* __restrict__ outh)
{
  const int s = blockIdx.x >> 6;
  const int g = blockIdx.x & 63;
  const int b = g >> 2;
  const int h = ((g & 3) << 8) + threadIdx.x;
  const int chan = b * H_ + h;
  float c = Cin[(size_t)s * BH_ + chan];
  const int t0 = s * TSEG_;
  for (int t = 0; t < TSEG_; ++t) {
    const size_t zi = (size_t)(t0 + t) * BH_ + chan;
    const float z = (float)zws[zi];
    const float f = (float)fws[zi];
    c = f * c + (1.f - f) * z;
    const size_t oi = ((size_t)b * T_ + (t0 + t)) * H_ + h;
    outc[oi] = c;
    const float o = outh[oi];      // O stored here by the GEMM epilogue
    outh[oi] = o * c;
  }
}

// ---------------------------------------------------------------------------
extern "C" void kernel_launch(void* const* d_in, const int* in_sizes, int n_in,
                              void* d_out, int out_size, void* d_ws, size_t ws_size,
                              hipStream_t stream) {
  const float* x    = (const float*)d_in[0];   // [B,T,D] fp32
  const float* w    = (const float*)d_in[1];   // [3H,D,2] fp32
  const float* bias = (const float*)d_in[2];   // [3H] fp32
  // d_in[3] = layer_idx (ignored: layer 0 path)

  char* ws = (char*)d_ws;
  __bf16* wb  = (__bf16*)(ws + WB_OFF);
  __bf16* ap  = (__bf16*)(ws + AP_OFF);
  __bf16* zws = (__bf16*)(ws + Z_OFF);
  __bf16* fws = (__bf16*)(ws + F_OFF);
  float*  P   = (float*)(ws + P_OFF);
  float*  C   = (float*)(ws + C_OFF);

  float* outc = (float*)d_out;
  float* outh = outc + (size_t)B_ * T_ * H_;

  pack_w_kernel<<<(NT_ * D_) / 256, 256, 0, stream>>>(w, wb);
  pack_a_kernel<<<(M_ * D_ / 8) / 256, 256, 0, stream>>>(x, ap);
  gemm_gates_kernel<<<dim3(NT_ / 128, M_ / 128), 256, 0, stream>>>(ap, wb, bias, zws, fws, outh);
  scan_seg_reduce<<<SEG_ * 64, 256, 0, stream>>>(zws, fws, P, C);
  scan_carry<<<BH_ / 256, 256, 0, stream>>>(P, C);
  scan_final<<<SEG_ * 64, 256, 0, stream>>>(zws, fws, C, outc, outh);
}
